// eViT_43843026158075
// MI455X (gfx1250) — matrix operand order repaired and checked by reading, not compiled, not run on hardware
//
#include <hip/hip_runtime.h>
#include <hip/hip_bf16.h>
#include <math.h>

#define N_NODES   100000
#define N_EDGES   1600000
#define N_PATCHES 256
#define D3        32
#define INNER     512
#define HEADS     8
#define DHEAD     64

static_assert((N_EDGES & 15) == 0, "edge tiles assumed full");

typedef __attribute__((ext_vector_type(16))) _Float16 v16h;
typedef __attribute__((ext_vector_type(8)))  float    v8f;

// ---- CDNA5 WMMA f16 lane layouts (ISA 7.12.2, wave32) ----
// A (16x32 MxK): lane L holds row M=L&15; half h -> K = (h&7) + 16*(h>>3) + 8*(L>>4)
// B (32x16 KxN): lane L holds col N=L&15; half h -> K = (h&7) + 16*(h>>3) + 8*(L>>4)
// C/D (16x16 f32): elem f -> M = f + 8*(L>>4), N = L&15
__device__ __forceinline__ int ka_map(int h, int g) { return (h & 7) + ((h >> 3) << 4) + (g << 3); }
__device__ __forceinline__ int kb_map(int h, int g) { return (h & 7) + ((h >> 3) << 4) + (g << 3); }

__device__ __forceinline__ v8f wmma_f16(v16h a, v16h b, v8f c) {
  return __builtin_amdgcn_wmma_f32_16x16x32_f16(false, a, false, b, (short)0, c, false, false);
}

__device__ __forceinline__ void lds_fence() {
  asm volatile("s_wait_dscnt 0" ::: "memory");
}

// ------------------------------------------------------------------
__global__ void fill_u32_kernel(unsigned* __restrict__ p, unsigned v, size_t n) {
  size_t i = (size_t)blockIdx.x * blockDim.x + threadIdx.x;
  if (i < n) p[i] = v;
}

__global__ void encode_kernel(const float* __restrict__ x, const float* __restrict__ w,
                              const float* __restrict__ b, float* __restrict__ h0,
                              _Float16* __restrict__ h016) {
  int n = blockIdx.x * blockDim.x + threadIdx.x;
  if (n >= N_NODES) return;
  float xv = x[n];
#pragma unroll
  for (int j = 0; j < 8; ++j) {
    float v = xv * w[j] + b[j];
    h0[n * 8 + j] = v;
    h016[n * 8 + j] = (_Float16)v;
  }
}

__global__ void count_kernel(const int* __restrict__ dst, float* __restrict__ cnt) {
  int e = blockIdx.x * blockDim.x + threadIdx.x;
  if (e < N_EDGES) atomicAdd(&cnt[dst[e]], 1.0f);
}

// ---- edge message pass: per 16-edge tile, msg = (basis (x) xs) @ W, WMMA f16 ----
// Gathers come from an f16 mirror of h: 16/32 contiguous bytes per edge row
// (b128 global loads -> b128 LDS stores), dst staged in LDS, edge_attr as float2.
template <int DIN, int DOUT>
__global__ void edge_spline_kernel(const _Float16* __restrict__ hprev16,
                                   const int*      __restrict__ src,
                                   const int*      __restrict__ dst,
                                   const float2*   __restrict__ ea2,
                                   const float*    __restrict__ ws,
                                   float* __restrict__ agg,
                                   int n_tiles, int n_iter) {
  constexpr int K  = 4 * DIN;   // 32 or 64
  constexpr int KT = K / 32;    // 1 or 2
  constexpr int NT = DOUT / 16; // 1 or 2
  __shared__ _Float16 Wl[K * DOUT];
  __shared__ __align__(16) _Float16 xs[8][16][DIN];
  __shared__ float    bas[8][16][4];
  __shared__ int      dstl[8][16];

  const int tid = threadIdx.x;
  for (int i = tid; i < K * DOUT; i += 256) Wl[i] = (_Float16)ws[i];
  __syncthreads();

  const int wave = tid >> 5, lane = tid & 31;
  const int g = lane >> 4, r = lane & 15;

  // Weights operand is loop-invariant: hoist B tiles into VGPRs.
  v16h B[NT][KT];
#pragma unroll
  for (int nt = 0; nt < NT; ++nt)
#pragma unroll
    for (int kt = 0; kt < KT; ++kt)
#pragma unroll
      for (int h = 0; h < 16; ++h)
        B[nt][kt][h] = Wl[(kt * 32 + kb_map(h, g)) * DOUT + nt * 16 + r];

  const int gw = blockIdx.x * 8 + wave;
  const int tw = gridDim.x * 8;

  for (int it = 0; it < n_iter; ++it) {
    int tile = it * tw + gw;
    if (tile >= n_tiles) continue;   // wave-uniform branch, EXEC stays full
    int e0 = tile * 16;

    // lanes 0..15 stage one edge each (wave-private LDS, no block barrier needed)
    if (lane < 16) {
      int e = e0 + lane;
      int s = src[e];
      dstl[wave][lane] = dst[e];
      float2 u = ea2[e];
      bas[wave][lane][0] = (1.f - u.x) * (1.f - u.y);
      bas[wave][lane][1] = (1.f - u.x) * u.y;
      bas[wave][lane][2] = u.x * (1.f - u.y);
      bas[wave][lane][3] = u.x * u.y;
      const uint4* row = (const uint4*)(hprev16 + (size_t)s * DIN);
      uint4* xrow = (uint4*)&xs[wave][lane][0];
      xrow[0] = row[0];                       // 8 halves
      if constexpr (DIN == 16) xrow[1] = row[1];
    }
    lds_fence();  // same-wave LDS is in-order; wait for staging stores

    // Build A tile: z[K] = basis[K/DIN] * xs[K%DIN]
    v16h A[KT];
#pragma unroll
    for (int kt = 0; kt < KT; ++kt)
#pragma unroll
      for (int h = 0; h < 16; ++h) {
        int kk = kt * 32 + ka_map(h, g);
        A[kt][h] = (_Float16)(bas[wave][r][kk / DIN] * (float)xs[wave][r][kk % DIN]);
      }

#pragma unroll
    for (int nt = 0; nt < NT; ++nt) {
      v8f D = {};
#pragma unroll
      for (int kt = 0; kt < KT; ++kt) D = wmma_f16(A[kt], B[nt][kt], D);
      // scatter-add into agg[dst] (L2-resident fp32 atomics); tiles are always full
#pragma unroll
      for (int f = 0; f < 8; ++f)
        atomicAdd(&agg[(size_t)dstl[wave][f + 8 * g] * DOUT + nt * 16 + r], D[f]);
    }
  }
}

// ---- node update: h' = elu(agg/max(cnt,1) + h@wr + b), f32 + f16 mirror ----
template <int DIN, int DOUT>
__global__ void node_update_kernel(const float* __restrict__ hprev, const float* __restrict__ agg,
                                   const float* __restrict__ cnt, const float* __restrict__ wr,
                                   const float* __restrict__ b, float* __restrict__ hnext,
                                   _Float16* __restrict__ hnext16) {
  __shared__ float wl[DIN * DOUT];
  __shared__ float bl[DOUT];
  int tid = threadIdx.x;
  for (int i = tid; i < DIN * DOUT; i += 256) wl[i] = wr[i];
  for (int i = tid; i < DOUT; i += 256) bl[i] = b[i];
  __syncthreads();
  int n = blockIdx.x * blockDim.x + tid;
  if (n >= N_NODES) return;
  float hi[DIN];
#pragma unroll
  for (int i = 0; i < DIN; ++i) hi[i] = hprev[(size_t)n * DIN + i];
  float inv = 1.f / fmaxf(cnt[n], 1.f);
#pragma unroll
  for (int j = 0; j < DOUT; ++j) {
    float a = bl[j] + agg[(size_t)n * DOUT + j] * inv;
#pragma unroll
    for (int i = 0; i < DIN; ++i) a += hi[i] * wl[i * DOUT + j];
    float v = a > 0.f ? a : (expf(a) - 1.f);
    hnext[(size_t)n * DOUT + j] = v;
    hnext16[(size_t)n * DOUT + j] = (_Float16)v;
  }
}

// ---- segment-max pooling via order-preserving uint atomicMax ----
__device__ __forceinline__ unsigned ord_f32(float f) {
  unsigned u = __float_as_uint(f);
  return (u & 0x80000000u) ? ~u : (u | 0x80000000u);
}

__global__ void pool_kernel(const float* __restrict__ h3, const int* __restrict__ npm,
                            unsigned* __restrict__ pu) {
  int n = blockIdx.x * blockDim.x + threadIdx.x;
  if (n >= N_NODES) return;
  int p = npm[n];
#pragma unroll
  for (int j = 0; j < D3; ++j)
    atomicMax(&pu[p * D3 + j], ord_f32(h3[(size_t)n * D3 + j]));
}

__global__ void pool_fix_kernel(const unsigned* __restrict__ pu, float* __restrict__ pf) {
  int i = blockIdx.x * blockDim.x + threadIdx.x;
  if (i >= N_PATCHES * D3) return;
  unsigned o = pu[i];
  unsigned b = (o & 0x80000000u) ? (o & 0x7FFFFFFFu) : ~o;
  float f = __uint_as_float(b);
  pf[i] = isfinite(f) ? f : 0.f;  // empty segment (never updated) -> NaN -> 0, matches ref
}

// ---- qkv = pooled @ w_qkv : one WMMA per 16x16 tile (K = 32 exactly) ----
__global__ void qkv_kernel(const float* __restrict__ pf, const float* __restrict__ w_qkv,
                           _Float16* __restrict__ qkvh) {
  int bm = blockIdx.x / 96, bn = blockIdx.x % 96;
  int lane = threadIdx.x, g = lane >> 4, r = lane & 15;
  v16h A, B;
#pragma unroll
  for (int h = 0; h < 16; ++h) {
    int kk = ka_map(h, g);
    A[h] = (_Float16)pf[(bm * 16 + r) * D3 + kk];
  }
#pragma unroll
  for (int h = 0; h < 16; ++h) {
    int kk = kb_map(h, g);
    B[h] = (_Float16)w_qkv[kk * (3 * INNER) + bn * 16 + r];
  }
  v8f D = {};
  D = wmma_f16(A, B, D);
#pragma unroll
  for (int f = 0; f < 8; ++f)
    qkvh[(size_t)(bm * 16 + f + 8 * g) * (3 * INNER) + bn * 16 + r] = (_Float16)D[f];
}

// ---- per (head, 16-query tile): scores + softmax + P@V, single wave ----
__global__ void attn_kernel(const _Float16* __restrict__ qkvh, float* __restrict__ o) {
  __shared__ float    S[16][256];
  __shared__ _Float16 P[16][256];
  int head = blockIdx.x >> 4, qt = blockIdx.x & 15;
  int lane = threadIdx.x, g = lane >> 4, r = lane & 15;
  int qbase = qt * 16;
  const int ROW = 3 * INNER;

  // Q tile operand (row = query, K = head dim 0..63)
  v16h Aq[2];
#pragma unroll
  for (int kt = 0; kt < 2; ++kt)
#pragma unroll
    for (int h = 0; h < 16; ++h) {
      int kk = kt * 32 + ka_map(h, g);
      Aq[kt][h] = qkvh[(size_t)(qbase + r) * ROW + head * DHEAD + kk];
    }

  // scores S = Q K^T * scale
  for (int ct = 0; ct < 16; ++ct) {
    v8f D = {};
#pragma unroll
    for (int kt = 0; kt < 2; ++kt) {
      v16h B;
#pragma unroll
      for (int h = 0; h < 16; ++h) {
        int kk = kt * 32 + kb_map(h, g);   // B[K=dim][N=key] = K_mat[key][dim]
        B[h] = qkvh[(size_t)(ct * 16 + r) * ROW + INNER + head * DHEAD + kk];
      }
      D = wmma_f16(Aq[kt], B, D);
    }
#pragma unroll
    for (int f = 0; f < 8; ++f) S[f + 8 * g][ct * 16 + r] = D[f] * 0.125f; // 64^-0.5
  }
  lds_fence();

  // softmax over each of the 16 rows, whole wave cooperates
  for (int row = 0; row < 16; ++row) {
    float m = -INFINITY;
    for (int c = lane; c < 256; c += 32) m = fmaxf(m, S[row][c]);
#pragma unroll
    for (int off = 16; off > 0; off >>= 1) m = fmaxf(m, __shfl_xor(m, off, 32));
    float sum = 0.f;
    for (int c = lane; c < 256; c += 32) {
      float e = __expf(S[row][c] - m);
      S[row][c] = e;
      sum += e;
    }
#pragma unroll
    for (int off = 16; off > 0; off >>= 1) sum += __shfl_xor(sum, off, 32);
    float inv = 1.f / sum;
    for (int c = lane; c < 256; c += 32) P[row][c] = (_Float16)(S[row][c] * inv);
  }
  lds_fence();

  // O = P @ V  (K = 256 -> 8 WMMAs per 16x16 output tile)
#pragma unroll
  for (int nt = 0; nt < 4; ++nt) {
    v8f D = {};
    for (int kt = 0; kt < 8; ++kt) {
      v16h A, B;
#pragma unroll
      for (int h = 0; h < 16; ++h) {
        int kk = kt * 32 + ka_map(h, g);
        A[h] = P[r][kk];
      }
#pragma unroll
      for (int h = 0; h < 16; ++h) {
        int kk = kt * 32 + kb_map(h, g);   // B[K=key][N=dim] = V[key][dim]
        B[h] = qkvh[(size_t)kk * ROW + 2 * INNER + head * DHEAD + nt * 16 + r];
      }
      D = wmma_f16(A, B, D);
    }
#pragma unroll
    for (int f = 0; f < 8; ++f)
      o[(size_t)(qbase + f + 8 * g) * INNER + head * DHEAD + nt * 16 + r] = D[f];
  }
}

__global__ void colmean_kernel(const float* __restrict__ o, float* __restrict__ mean) {
  int j = blockIdx.x * blockDim.x + threadIdx.x;
  if (j >= INNER) return;
  float s = 0.f;
  for (int p = 0; p < N_PATCHES; ++p) s += o[(size_t)p * INNER + j];
  mean[j] = s * (1.f / N_PATCHES);
}

__global__ void outproj_kernel(const float* __restrict__ mean, const float* __restrict__ w_out,
                               const float* __restrict__ b_out, float* __restrict__ out) {
  int j = threadIdx.x;
  if (j >= D3) return;
  float acc = b_out[j];
  for (int i = 0; i < INNER; ++i) acc += mean[i] * w_out[i * D3 + j];
  out[j] = acc;
}

// ------------------------------------------------------------------
extern "C" void kernel_launch(void* const* d_in, const int* in_sizes, int n_in,
                              void* d_out, int out_size, void* d_ws, size_t ws_size,
                              hipStream_t stream) {
  (void)in_sizes; (void)n_in; (void)out_size; (void)ws_size;
  const float* x     = (const float*)d_in[0];
  const int*   eidx  = (const int*)d_in[1];
  const float* ea    = (const float*)d_in[2];
  const int*   npm   = (const int*)d_in[3];
  const float* w_enc = (const float*)d_in[4];
  const float* b_enc = (const float*)d_in[5];
  const float* ws1   = (const float*)d_in[6];
  const float* wr1   = (const float*)d_in[7];
  const float* b1    = (const float*)d_in[8];
  const float* ws2   = (const float*)d_in[9];
  const float* wr2   = (const float*)d_in[10];
  const float* b2    = (const float*)d_in[11];
  const float* ws3   = (const float*)d_in[12];
  const float* wr3   = (const float*)d_in[13];
  const float* b3    = (const float*)d_in[14];
  const float* w_qkv = (const float*)d_in[15];
  const float* w_out = (const float*)d_in[16];
  const float* b_out = (const float*)d_in[17];
  const int* src = eidx;
  const int* dst = eidx + N_EDGES;
  const float2* ea2 = (const float2*)ea;
  float* out = (float*)d_out;

  // workspace layout (floats; all segments 16B-aligned)
  float*    wsf  = (float*)d_ws;
  float*    hA   = wsf;                                  // 100000*32
  float*    hB   = hA + (size_t)N_NODES * 32;            // 100000*32
  float*    agg  = hB + (size_t)N_NODES * 32;            // 100000*32
  float*    cnt  = agg + (size_t)N_NODES * 32;           // 100000
  unsigned* pu   = (unsigned*)(cnt + N_NODES);           // 256*32
  float*    pf   = (float*)(pu + N_PATCHES * D3);        // 256*32
  _Float16* qkvh = (_Float16*)(pf + N_PATCHES * D3);     // 256*1536 halves
  float*    of   = (float*)(qkvh + (size_t)N_PATCHES * 3 * INNER); // 256*512
  float*    mean = of + (size_t)N_PATCHES * INNER;       // 512
  _Float16* hh16 = (_Float16*)(mean + INNER);            // 100000*32 halves (f16 mirror)

  auto fill = [&](unsigned* p, unsigned v, size_t n) {
    fill_u32_kernel<<<(unsigned)((n + 255) / 256), 256, 0, stream>>>(p, v, n);
  };

  const int n_tiles = N_EDGES / 16;                      // 100000 (all tiles full)
  const int edge_blocks = 1024;                          // 8192 waves
  const int n_iter = (n_tiles + edge_blocks * 8 - 1) / (edge_blocks * 8);
  const int node_blocks = (N_NODES + 255) / 256;

  // in-degree counts (shared by all 3 blocks)
  fill((unsigned*)cnt, 0u, N_NODES);
  count_kernel<<<(N_EDGES + 255) / 256, 256, 0, stream>>>(dst, cnt);

  // encoder (writes f32 + f16 mirror)
  encode_kernel<<<node_blocks, 256, 0, stream>>>(x, w_enc, b_enc, hA, hh16);

  // block 1: 8 -> 16   (hA -> hB)
  fill((unsigned*)agg, 0u, (size_t)N_NODES * 16);
  edge_spline_kernel<8, 16><<<edge_blocks, 256, 0, stream>>>(hh16, src, dst, ea2, ws1, agg, n_tiles, n_iter);
  node_update_kernel<8, 16><<<node_blocks, 256, 0, stream>>>(hA, agg, cnt, wr1, b1, hB, hh16);

  // block 2: 16 -> 16  (hB -> hA)
  fill((unsigned*)agg, 0u, (size_t)N_NODES * 16);
  edge_spline_kernel<16, 16><<<edge_blocks, 256, 0, stream>>>(hh16, src, dst, ea2, ws2, agg, n_tiles, n_iter);
  node_update_kernel<16, 16><<<node_blocks, 256, 0, stream>>>(hB, agg, cnt, wr2, b2, hA, hh16);

  // block 3: 16 -> 32  (hA -> hB)
  fill((unsigned*)agg, 0u, (size_t)N_NODES * 32);
  edge_spline_kernel<16, 32><<<edge_blocks, 256, 0, stream>>>(hh16, src, dst, ea2, ws3, agg, n_tiles, n_iter);
  node_update_kernel<16, 32><<<node_blocks, 256, 0, stream>>>(hA, agg, cnt, wr3, b3, hB, hh16);

  // segment-max pooling over patches
  fill(pu, 0u, (size_t)N_PATCHES * D3);
  pool_kernel<<<node_blocks, 256, 0, stream>>>(hB, npm, pu);
  pool_fix_kernel<<<(N_PATCHES * D3 + 255) / 256, 256, 0, stream>>>(pu, pf);

  // attention (WMMA f16): qkv, per-head scores/softmax/PV, then mean+project
  qkv_kernel<<<16 * 96, 32, 0, stream>>>(pf, w_qkv, qkvh);
  attn_kernel<<<HEADS * 16, 32, 0, stream>>>(qkvh, of);
  colmean_kernel<<<(INNER + 255) / 256, 256, 0, stream>>>(of, mean);
  outproj_kernel<<<1, 32, 0, stream>>>(mean, w_out, b_out, out);
}
